// CPUEfficientCausalAttention_83846351552988
// MI455X (gfx1250) — matrix operand order repaired and checked by reading, not compiled, not run on hardware
//
#include <hip/hip_runtime.h>
#include <hip/hip_bf16.h>

// ---------------------------------------------------------------------------
// Causal MHA for MI455X (gfx1250, wave32, WMMA f16 -> f32 accumulate).
// - All intermediates f16 in workspace (~48MB) => fully L2-resident (192MB L2)
// - GEMMs register-blocked 32Mx64N per wave: 2 A-frags x 4 B-frags = 8 WMMAs
//   per 32-K step; B-frags loaded into distinct registers ahead of the WMMAs
//   so loads pipeline (avoid s_wait_loadcnt 0 before every WMMA).
// - V stored transposed [b][h][d][s] so the P@V B-fragment is a contiguous
//   32B load (same shape as the K fragment) -- no per-element gather.
// ---------------------------------------------------------------------------

#define HIDDEN   1024
#define HEADS    16
#define HEAD_DIM 64
#define BATCH    2
#define SEQ      2048
#define MROWS    (BATCH * SEQ)       // 4096
#define QKV_N    (3 * HIDDEN)        // 3072

typedef _Float16 v16h __attribute__((ext_vector_type(16)));
typedef _Float16 v8h  __attribute__((ext_vector_type(8)));
typedef float    v8f  __attribute__((ext_vector_type(8)));

static __device__ __forceinline__ v8f zero8() {
  v8f z;
#pragma unroll
  for (int i = 0; i < 8; ++i) z[i] = 0.0f;
  return z;
}

static __device__ __forceinline__ v16h mk16(v8h lo, v8h hi) {
  v16h r;
#pragma unroll
  for (int i = 0; i < 8; ++i) { r[i] = lo[i]; r[i + 8] = hi[i]; }
  return r;
}

#define WMMA_F16(A, B, C)                                                    \
  __builtin_amdgcn_wmma_f32_16x16x32_f16(false, (A), false, (B), (short)0,   \
                                         (C), false, false)

// ------------------------------- convert -----------------------------------
__global__ void k_f32_to_f16(const float* __restrict__ in,
                             _Float16* __restrict__ out, int n) {
  int i = blockIdx.x * blockDim.x + threadIdx.x;
  if (i < n) out[i] = (_Float16)in[i];
}

// ------------------------------ QKV GEMM ------------------------------------
// C[m,n] = sum_k x16[m,k] * qkvw16[n,k]; one wave per 32Mx64N tile.
// Scatter: Q[b][h][s][hd]*scale, K[b][h][s][hd], V transposed [b][h][hd][s].
__global__ void k_gemm_qkv(const _Float16* __restrict__ A,  // [4096][1024]
                           const _Float16* __restrict__ W,  // [3072][1024]
                           _Float16* __restrict__ Qh,
                           _Float16* __restrict__ Kh,
                           _Float16* __restrict__ Vt) {
  const int tM = blockIdx.x * 32;
  const int tN = blockIdx.y * 64;
  const int lane = threadIdx.x & 31;
  const int half = lane >> 4;
  const int l16  = lane & 15;

  const _Float16* arow0 = A + (size_t)(tM + l16) * HIDDEN;
  const _Float16* arow1 = A + (size_t)(tM + 16 + l16) * HIDDEN;
  const _Float16* brow  = W + (size_t)(tN + l16) * HIDDEN + 16 * half;

  v8f acc[2][4];
#pragma unroll
  for (int m = 0; m < 2; ++m)
#pragma unroll
    for (int t = 0; t < 4; ++t) acc[m][t] = zero8();

  for (int kk = 0; kk < HIDDEN; kk += 32) {
    // A-frags (16x32 f16): lanes 0-15 hold K kk+0..15,
    //                      lanes 16-31 hold K kk+16..31.
    const v16h a0 = *(const v16h*)(arow0 + kk + 16 * half);
    const v16h a1 = *(const v16h*)(arow1 + kk + 16 * half);
    // B-frags into distinct registers first, then the 8 WMMAs.
    v16h b[4];
#pragma unroll
    for (int nT = 0; nT < 4; ++nT)
      b[nT] = *(const v16h*)(brow + (size_t)(16 * nT) * HIDDEN + kk);
#pragma unroll
    for (int nT = 0; nT < 4; ++nT) {
      acc[0][nT] = WMMA_F16(a0, b[nT], acc[0][nT]);
      acc[1][nT] = WMMA_F16(a1, b[nT], acc[1][nT]);
    }
  }

  // C layout: value[r] at row tM + 16*mT + r + 8*half, col tN + 16*nT + l16.
  const int bb  = tM >> 11;                       // / SEQ (tile within batch)
#pragma unroll
  for (int mT = 0; mT < 2; ++mT) {
    const int ss0 = (tM & (SEQ - 1)) + 16 * mT + 8 * half;
#pragma unroll
    for (int nT = 0; nT < 4; ++nT) {
      const int nG = tN + 16 * nT + l16;
      const int which = nG >> 10;                 // 0=q 1=k 2=v
      const int hh = (nG >> 6) & (HEADS - 1);
      const int dd = nG & (HEAD_DIM - 1);
      if (which == 2) {
        _Float16* p =
            Vt + ((size_t)(bb * HEADS + hh) * HEAD_DIM + dd) * SEQ + ss0;
#pragma unroll
        for (int r = 0; r < 8; ++r) p[r] = (_Float16)acc[mT][nT][r];
      } else {
        const float mul = which ? 1.0f : 0.125f;  // HEAD_DIM^-0.5 on Q
        _Float16* p = (which ? Kh : Qh) +
                      ((size_t)(bb * HEADS + hh) * SEQ + ss0) * HEAD_DIM + dd;
#pragma unroll
        for (int r = 0; r < 8; ++r)
          p[(size_t)r * HEAD_DIM] = (_Float16)(acc[mT][nT][r] * mul);
      }
    }
  }
}

// --------------------------- flash attention --------------------------------
// One wave32 per (16-query tile, head, batch). Online softmax, causal.
__global__ void k_attn(const _Float16* __restrict__ Qh,
                       const _Float16* __restrict__ Kh,
                       const _Float16* __restrict__ Vt,
                       _Float16* __restrict__ Oh) {  // [b][s][h*64+d] f16
  const int lane = threadIdx.x & 31;
  const int half = lane >> 4;
  const int l16  = lane & 15;
  const int qTile = blockIdx.x;
  const int h = blockIdx.y;
  const int b = blockIdx.z;

  const size_t hb = (size_t)(b * HEADS + h) * SEQ * HEAD_DIM;
  const _Float16* Qb  = Qh + hb;
  const _Float16* Kb  = Kh + hb;
  const _Float16* Vtb = Vt + hb;   // [hd][s]

  const int q0 = qTile * 16;

  // Preload the two Q A-frags covering hd=64 (K-steps kk=0,32).
  const _Float16* qr = Qb + (size_t)(q0 + l16) * HEAD_DIM;
  const v16h aq0 = *(const v16h*)(qr + 16 * half);
  const v16h aq1 = *(const v16h*)(qr + 32 + 16 * half);

  v8f oacc[4];
#pragma unroll
  for (int t = 0; t < 4; ++t) oacc[t] = zero8();
  float mrow[8], lrow[8];
#pragma unroll
  for (int r = 0; r < 8; ++r) { mrow[r] = -1e30f; lrow[r] = 0.0f; }

  __shared__ __align__(16) _Float16 pLds[16 * 32];

  const int qEnd = q0 + 15;
  for (int k0 = 0; k0 <= qEnd; k0 += 32) {
    // ---- scores for 32 keys: two 16x16 tiles ----
    v8f sc[2];
#pragma unroll
    for (int t = 0; t < 2; ++t) {
      const _Float16* kr = Kb + (size_t)(k0 + 16 * t + l16) * HEAD_DIM;
      v8f s = zero8();
      s = WMMA_F16(aq0, *(const v16h*)(kr + 16 * half), s);
      s = WMMA_F16(aq1, *(const v16h*)(kr + 32 + 16 * half), s);
      sc[t] = s;
    }

    // ---- causal mask + online softmax (C layout: row=r+8*half, col=l16) ----
    float pv0[8], pv1[8], alpha[8];
#pragma unroll
    for (int r = 0; r < 8; ++r) {
      const int qrow = q0 + r + 8 * half;
      float s0 = (k0 + l16 <= qrow) ? sc[0][r] : -1e30f;
      float s1 = (k0 + 16 + l16 <= qrow) ? sc[1][r] : -1e30f;
      float mx = fmaxf(s0, s1);
      mx = fmaxf(mx, __shfl_xor(mx, 1, 32));
      mx = fmaxf(mx, __shfl_xor(mx, 2, 32));
      mx = fmaxf(mx, __shfl_xor(mx, 4, 32));
      mx = fmaxf(mx, __shfl_xor(mx, 8, 32));
      const float mnew = fmaxf(mrow[r], mx);
      alpha[r] = __expf(mrow[r] - mnew);
      const float p0 = __expf(s0 - mnew);
      const float p1 = __expf(s1 - mnew);
      float rs = p0 + p1;
      rs += __shfl_xor(rs, 1, 32);
      rs += __shfl_xor(rs, 2, 32);
      rs += __shfl_xor(rs, 4, 32);
      rs += __shfl_xor(rs, 8, 32);
      lrow[r] = lrow[r] * alpha[r] + rs;
      mrow[r] = mnew;
      pv0[r] = p0; pv1[r] = p1;
    }

    // rescale accumulators, stage P in LDS (row-major 16x32)
#pragma unroll
    for (int r = 0; r < 8; ++r) {
      oacc[0][r] *= alpha[r]; oacc[1][r] *= alpha[r];
      oacc[2][r] *= alpha[r]; oacc[3][r] *= alpha[r];
      const int row = r + 8 * half;
      pLds[row * 32 + l16]      = (_Float16)pv0[r];
      pLds[row * 32 + 16 + l16] = (_Float16)pv1[r];
    }
    __syncthreads();

    // re-fragment P as A-matrix 16x32
    const _Float16* pr = pLds + l16 * 32;
    const v16h pa = mk16(*(const v8h*)(pr + 16 * half),
                         *(const v8h*)(pr + 16 * half + 8));
    __syncthreads();

    // ---- P @ V: four 16x16 N tiles over hd=64; B-frag element e is
    // Vt[n][k0+16*half+e] -> one contiguous 32B load per tile ----
    v16h vb[4];
#pragma unroll
    for (int nT = 0; nT < 4; ++nT)
      vb[nT] = *(const v16h*)(Vtb + (size_t)(nT * 16 + l16) * SEQ + k0 +
                              16 * half);
#pragma unroll
    for (int nT = 0; nT < 4; ++nT) oacc[nT] = WMMA_F16(pa, vb[nT], oacc[nT]);
  }

  // normalize + store to O16 [b][s][h*64+d]
#pragma unroll
  for (int nT = 0; nT < 4; ++nT) {
#pragma unroll
    for (int r = 0; r < 8; ++r) {
      const int srow = q0 + r + 8 * half;
      const int d = nT * 16 + l16;
      Oh[((size_t)b * SEQ + srow) * HIDDEN + h * HEAD_DIM + d] =
          (_Float16)(oacc[nT][r] / lrow[r]);
    }
  }
}

// ------------------------------ out GEMM ------------------------------------
// Y[m,n](f32) = sum_k O16[m,k] * ow16[n,k]; one wave per 32Mx64N tile.
__global__ void k_gemm_out(const _Float16* __restrict__ A,  // [4096][1024]
                           const _Float16* __restrict__ W,  // [1024][1024]
                           float* __restrict__ Y) {
  const int tM = blockIdx.x * 32;
  const int tN = blockIdx.y * 64;
  const int lane = threadIdx.x & 31;
  const int half = lane >> 4;
  const int l16  = lane & 15;

  const _Float16* arow0 = A + (size_t)(tM + l16) * HIDDEN;
  const _Float16* arow1 = A + (size_t)(tM + 16 + l16) * HIDDEN;
  const _Float16* brow  = W + (size_t)(tN + l16) * HIDDEN + 16 * half;

  v8f acc[2][4];
#pragma unroll
  for (int m = 0; m < 2; ++m)
#pragma unroll
    for (int t = 0; t < 4; ++t) acc[m][t] = zero8();

  for (int kk = 0; kk < HIDDEN; kk += 32) {
    const v16h a0 = *(const v16h*)(arow0 + kk + 16 * half);
    const v16h a1 = *(const v16h*)(arow1 + kk + 16 * half);
    v16h b[4];
#pragma unroll
    for (int nT = 0; nT < 4; ++nT)
      b[nT] = *(const v16h*)(brow + (size_t)(16 * nT) * HIDDEN + kk);
#pragma unroll
    for (int nT = 0; nT < 4; ++nT) {
      acc[0][nT] = WMMA_F16(a0, b[nT], acc[0][nT]);
      acc[1][nT] = WMMA_F16(a1, b[nT], acc[1][nT]);
    }
  }

#pragma unroll
  for (int mT = 0; mT < 2; ++mT) {
    float* yp = Y + (size_t)(tM + 16 * mT + 8 * half) * HIDDEN + tN + l16;
#pragma unroll
    for (int nT = 0; nT < 4; ++nT)
#pragma unroll
      for (int r = 0; r < 8; ++r)
        yp[(size_t)r * HIDDEN + 16 * nT] = acc[mT][nT][r];
  }
}

// ------------------------------- launcher -----------------------------------
extern "C" void kernel_launch(void* const* d_in, const int* in_sizes, int n_in,
                              void* d_out, int out_size, void* d_ws,
                              size_t ws_size, hipStream_t stream) {
  const float* x     = (const float*)d_in[0];   // [2,2048,1024]
  const float* qkv_w = (const float*)d_in[1];   // [3072,1024]
  const float* out_w = (const float*)d_in[2];   // [1024,1024]
  float* y = (float*)d_out;

  char* ws = (char*)d_ws;
  const size_t MB = 1024 * 1024;
  _Float16* x16  = (_Float16*)(ws + 0 * MB);    // 8 MB
  _Float16* qw16 = (_Float16*)(ws + 8 * MB);    // 6 MB
  _Float16* ow16 = (_Float16*)(ws + 14 * MB);   // 2 MB
  _Float16* Qh   = (_Float16*)(ws + 16 * MB);   // 8 MB
  _Float16* Kh   = (_Float16*)(ws + 24 * MB);   // 8 MB
  _Float16* Vt   = (_Float16*)(ws + 32 * MB);   // 8 MB  (transposed V)
  _Float16* Oh   = (_Float16*)(ws + 40 * MB);   // 8 MB

  const int nx = MROWS * HIDDEN;       // 4194304
  const int nw = QKV_N * HIDDEN;       // 3145728
  const int no = HIDDEN * HIDDEN;      // 1048576
  k_f32_to_f16<<<(nx + 255) / 256, 256, 0, stream>>>(x, x16, nx);
  k_f32_to_f16<<<(nw + 255) / 256, 256, 0, stream>>>(qkv_w, qw16, nw);
  k_f32_to_f16<<<(no + 255) / 256, 256, 0, stream>>>(out_w, ow16, no);

  k_gemm_qkv<<<dim3(MROWS / 32, QKV_N / 64), 32, 0, stream>>>(
      x16, qw16, Qh, Kh, Vt);

  k_attn<<<dim3(SEQ / 16, HEADS, BATCH), 32, 0, stream>>>(Qh, Kh, Vt, Oh);

  k_gemm_out<<<dim3(MROWS / 32, HIDDEN / 64), 32, 0, stream>>>(Oh, ow16, y);
}
